// SelfAttention_5763846111929
// MI455X (gfx1250) — matrix operand order repaired and checked by reading, not compiled, not run on hardware
//
#include <hip/hip_runtime.h>
#include <hip/hip_bf16.h>

// MI455X / gfx1250, wave32. Flash-attention with v_wmma_f32_16x16x32_f16 and
// double-buffered GLOBAL_LOAD_ASYNC_TO_LDS_B128 (ASYNCcnt) K/V staging.
//
// Shapes: B=4, C=256, Cq=64, N=H*W=4096.
// ws layout (f16): qT [B][N][64] | kT [B][N][64] | v [B][256][N]  (~12.6 MB)

typedef __attribute__((ext_vector_type(16))) _Float16 v16h;
typedef __attribute__((ext_vector_type(8)))  float    v8f;
typedef __attribute__((ext_vector_type(4)))  int      v4i;

union H16 { v16h v; _Float16 h[16]; uint4 u4[2]; };
union H8  { uint4 u; _Float16 h[8]; };

__device__ __forceinline__ v8f wmma_f16(const H16& a, const H16& b, v8f c) {
    return __builtin_amdgcn_wmma_f32_16x16x32_f16(false, a.v, false, b.v,
                                                  (short)0, c, false, false);
}

#define NPIX 4096
#define CDIM 256
#define CQ   64

#if __has_builtin(__builtin_amdgcn_global_load_async_to_lds_b128)
#define HAVE_ASYNC_LDS 1
typedef __attribute__((address_space(1))) v4i as1_v4i;   // global int4*
typedef __attribute__((address_space(3))) v4i as3_v4i;   // LDS int4*
#endif

__device__ __forceinline__ void wait_async0() {
#if __has_builtin(__builtin_amdgcn_s_wait_asynccnt)
    __builtin_amdgcn_s_wait_asynccnt(0);
#else
    asm volatile("s_wait_asynccnt 0x0" ::: "memory");
#endif
}

// ---------------------------------------------------------------------------
// Kernel 1: fused QKV 1x1-conv projection (per-batch GEMMs), f32 -> f16 out.
// Grid: (N/64, B), block 256 (8 waves).
// Wave w owns row-tiles {3w,3w+1,3w+2} of the stacked O-space:
//   tiles 0..3 -> Wq (64 rows), 4..7 -> Wk, 8..23 -> Wv (256 rows).
// ---------------------------------------------------------------------------
#define XTP   264   // xt LDS pitch (halves), mult of 8, bank-spread
#define SQKP  136   // stageQK pitch
#define SVP   72    // stageV pitch

__global__ __launch_bounds__(256) void qkv_proj_kernel(
    const float* __restrict__ x,
    const float* __restrict__ Wq, const float* __restrict__ bq,
    const float* __restrict__ Wk, const float* __restrict__ bk,
    const float* __restrict__ Wv, const float* __restrict__ bv,
    _Float16* __restrict__ qT, _Float16* __restrict__ kT,
    _Float16* __restrict__ vF)
{
    const int tid = threadIdx.x;
    const int l   = tid & 31;
    const int w   = __builtin_amdgcn_readfirstlane(tid >> 5);  // wave-uniform
    const int ln  = l & 15,  hh = l >> 4;
    const int b   = blockIdx.y;
    const int n0  = blockIdx.x * 64;

    __shared__ _Float16 smem[27136];           // 54.3 KB, time-shared
    _Float16* xt   = smem;                     // [64][XTP]  (x tile, transposed)
    _Float16* stQK = smem;                     // [64][SQKP] (q cols 0..63, k 64..127)
    _Float16* stV  = smem + 64 * SQKP;         // [256][SVP]

    // ---- stage x[b][:, n0..n0+63] transposed into xt[n][c], f16 ----
    {
        const int ch = tid & 3;        // 16-float chunk of the row
        const int cr = tid >> 2;       // base channel
        for (int p = 0; p < 4; ++p) {
            const int c = cr + p * 64;
            const float4* xs =
                (const float4*)(x + ((size_t)b * CDIM + c) * NPIX + n0 + ch * 16);
            float4 v0 = xs[0], v1 = xs[1], v2 = xs[2], v3 = xs[3];
            float f[16] = { v0.x, v0.y, v0.z, v0.w, v1.x, v1.y, v1.z, v1.w,
                            v2.x, v2.y, v2.z, v2.w, v3.x, v3.y, v3.z, v3.w };
#pragma unroll
            for (int q = 0; q < 16; ++q)
                xt[(ch * 16 + q) * XTP + c] = (_Float16)f[q];
        }
    }
    __syncthreads();

    v8f acc[12];
#pragma unroll
    for (int t = 0; t < 12; ++t)
#pragma unroll
        for (int i = 0; i < 8; ++i) acc[t][i] = 0.0f;

    // ---- K loop over C=256 in steps of 32 ----
    for (int cb = 0; cb < 8; ++cb) {
        H16 bf[4];
#pragma unroll
        for (int ct = 0; ct < 4; ++ct) {
            const int n = ct * 16 + ln;
            const uint4* p = (const uint4*)(xt + n * XTP + cb * 32 + hh * 8);
            bf[ct].u4[0] = p[0];
            bf[ct].u4[1] = p[2];
        }
#pragma unroll
        for (int j = 0; j < 3; ++j) {
            const int rt = 3 * w + j;  // wave-uniform
            const float* Wp; int orow;
            if (rt < 4)      { Wp = Wq; orow = rt * 16; }
            else if (rt < 8) { Wp = Wk; orow = (rt - 4) * 16; }
            else             { Wp = Wv; orow = (rt - 8) * 16; }
            const float* wr = Wp + (size_t)(orow + ln) * CDIM + cb * 32 + hh * 8;
            const float4* w4 = (const float4*)wr;
            float4 a0 = w4[0], a1 = w4[1], a2 = w4[4], a3 = w4[5];
            H16 af;
            af.h[0]=(_Float16)a0.x;  af.h[1]=(_Float16)a0.y;
            af.h[2]=(_Float16)a0.z;  af.h[3]=(_Float16)a0.w;
            af.h[4]=(_Float16)a1.x;  af.h[5]=(_Float16)a1.y;
            af.h[6]=(_Float16)a1.z;  af.h[7]=(_Float16)a1.w;
            af.h[8]=(_Float16)a2.x;  af.h[9]=(_Float16)a2.y;
            af.h[10]=(_Float16)a2.z; af.h[11]=(_Float16)a2.w;
            af.h[12]=(_Float16)a3.x; af.h[13]=(_Float16)a3.y;
            af.h[14]=(_Float16)a3.z; af.h[15]=(_Float16)a3.w;
#pragma unroll
            for (int ct = 0; ct < 4; ++ct)
                acc[j * 4 + ct] = wmma_f16(af, bf[ct], acc[j * 4 + ct]);
        }
    }
    __syncthreads();   // xt no longer needed; reuse smem for staging

    // ---- bias + f16 convert + stage for coalesced stores ----
#pragma unroll
    for (int j = 0; j < 3; ++j) {
        const int rt = 3 * w + j;
        const float* bp; int orow, matoff;
        if (rt < 4)      { bp = bq; orow = rt * 16;       matoff = 0;  }
        else if (rt < 8) { bp = bk; orow = (rt - 4) * 16; matoff = 64; }
        else             { bp = bv; orow = (rt - 8) * 16; matoff = -1; }
#pragma unroll
        for (int ct = 0; ct < 4; ++ct) {
            const int n = ct * 16 + ln;
            if (matoff >= 0) {
                H8 t;
#pragma unroll
                for (int i = 0; i < 8; ++i)
                    t.h[i] = (_Float16)(acc[j * 4 + ct][i] + bp[orow + 8 * hh + i]);
                *(uint4*)&stQK[n * SQKP + matoff + orow + 8 * hh] = t.u;
            } else {
#pragma unroll
                for (int i = 0; i < 8; ++i) {
                    const int o = orow + 8 * hh + i;
                    stV[o * SVP + n] = (_Float16)(acc[j * 4 + ct][i] + bp[o]);
                }
            }
        }
    }
    __syncthreads();

    // ---- coalesced copy-out ----
    {
        const int row = tid >> 2, ch = tid & 3;    // 32B chunks
        {
            const uint4* sq = (const uint4*)&stQK[row * SQKP + ch * 16];
            uint4* dq = (uint4*)(qT + ((size_t)b * NPIX + n0 + row) * CQ + ch * 16);
            dq[0] = sq[0]; dq[1] = sq[1];
            const uint4* sk = (const uint4*)&stQK[row * SQKP + 64 + ch * 16];
            uint4* dk = (uint4*)(kT + ((size_t)b * NPIX + n0 + row) * CQ + ch * 16);
            dk[0] = sk[0]; dk[1] = sk[1];
        }
        for (int p = 0; p < 4; ++p) {
            const int o = row + p * 64;
            const uint4* sv = (const uint4*)&stV[o * SVP + ch * 16];
            uint4* dv = (uint4*)(vF + ((size_t)b * CDIM + o) * NPIX + n0 + ch * 16);
            dv[0] = sv[0]; dv[1] = sv[1];
        }
    }
}

// ---------------------------------------------------------------------------
// Kernel 2: flash attention. Grid (N/64, B), block 256 (8 waves).
// Wave w: row-block r = w&3 (16 query rows), channel half h = w>>2 (128 ch).
// Per m-step of 32 keys: 4 WMMA (S) + online softmax + 8 WMMA (P*V).
// K/V tiles double-buffered in LDS via async load-to-LDS (ASYNCcnt).
// ---------------------------------------------------------------------------
#define KTP 72   // kt pitch (halves)
#define VTP 40   // vt pitch
#define PTP 40   // pt pitch

__global__ __launch_bounds__(256) void flash_attn_kernel(
    const _Float16* __restrict__ qT, const _Float16* __restrict__ kT,
    const _Float16* __restrict__ vF, const float* __restrict__ x,
    const float* __restrict__ gamma, float* __restrict__ out)
{
    const int tid = threadIdx.x;
    const int l   = tid & 31;
    const int w   = __builtin_amdgcn_readfirstlane(tid >> 5);
    const int ln  = l & 15,  hh = l >> 4;
    const int r   = w & 3,   hch = w >> 2;
    const int b   = blockIdx.y;
    const int n0  = blockIdx.x * 64;

    __shared__ _Float16 kt2[2][32 * KTP];     //  9.2 KB: kt[m][c]
    __shared__ _Float16 vt2[2][256 * VTP];    // 41.0 KB: vt[c][m]
    __shared__ _Float16 pt[8 * 16 * PTP];     // 10.2 KB: per-wave P[n][m]

    // Loop-invariant Q A-fragments (Cq=64 -> two K=32 steps)
    H16 qa[2];
    {
        const int n = n0 + r * 16 + ln;
        const _Float16* qr = qT + ((size_t)b * NPIX + n) * CQ;
#pragma unroll
        for (int kk = 0; kk < 2; ++kk) {
            qa[kk].u4[0] = *(const uint4*)(qr + kk * 32 + hh * 8);
            qa[kk].u4[1] = *(const uint4*)(qr + kk * 32 + 16 + hh * 8);
        }
    }

    v8f acc[8];
#pragma unroll
    for (int ct = 0; ct < 8; ++ct)
#pragma unroll
        for (int i = 0; i < 8; ++i) acc[ct][i] = 0.0f;
    float rm[8], rs[8];
#pragma unroll
    for (int i = 0; i < 8; ++i) { rm[i] = -1e30f; rs[i] = 0.0f; }

    _Float16* pw = &pt[w * 16 * PTP];

#ifdef HAVE_ASYNC_LDS
    // Per-lane DMA assignments: kt = 256 chunks of 16B (1/lane),
    // vt = 1024 chunks of 16B (4/lane). Pointers advance per m-step.
    const int kidx = w * 32 + l;
    const int krow = kidx >> 3, kch = kidx & 7;
    const _Float16* gk = kT + ((size_t)b * NPIX + krow) * CQ + kch * 8;
    const int klds = krow * KTP + kch * 8;
    const _Float16* gv[4];
    int vlds[4];
#pragma unroll
    for (int q = 0; q < 4; ++q) {
        const int vidx = w * 128 + q * 32 + l;
        const int vc = vidx >> 2, vch = vidx & 3;
        gv[q]   = vF + ((size_t)b * CDIM + vc) * NPIX + vch * 8;
        vlds[q] = vc * VTP + vch * 8;
    }

#define ISSUE_FILL(bufsel)                                                     \
    do {                                                                       \
        __builtin_amdgcn_global_load_async_to_lds_b128(                        \
            (as1_v4i*)gk, (as3_v4i*)&kt2[bufsel][klds], 0, 0);                 \
        gk += (size_t)32 * CQ;                                                 \
        _Pragma("unroll")                                                      \
        for (int q = 0; q < 4; ++q) {                                          \
            __builtin_amdgcn_global_load_async_to_lds_b128(                    \
                (as1_v4i*)gv[q], (as3_v4i*)&vt2[bufsel][vlds[q]], 0, 0);       \
            gv[q] += 32;                                                       \
        }                                                                      \
    } while (0)

    ISSUE_FILL(0);
#endif

    for (int step = 0; step < NPIX / 32; ++step) {
#ifdef HAVE_ASYNC_LDS
        const int buf = step & 1;
        wait_async0();        // my async fills for `buf` have landed
        __syncthreads();      // everyone's fills landed; prev reads of buf^1 done
        if (step + 1 < NPIX / 32) ISSUE_FILL(buf ^ 1);   // overlaps compute below
        const _Float16* ktb = kt2[buf];
        const _Float16* vtb = vt2[buf];
#else
        const int m0 = step * 32;
        __syncthreads();
        {   // kt fill: 32 rows x 128B from kT (contiguous)
            const int row = tid >> 3, ch = tid & 7;
            *(uint4*)&kt2[0][row * KTP + ch * 8] =
                *(const uint4*)(kT + ((size_t)b * NPIX + m0 + row) * CQ + ch * 8);
        }
        {   // vt fill: 256 rows x 64B from v (contiguous)
            const int c = tid;
            const uint4* s = (const uint4*)(vF + ((size_t)b * CDIM + c) * NPIX + m0);
            uint4* d = (uint4*)&vt2[0][c * VTP];
            d[0] = s[0]; d[1] = s[1]; d[2] = s[2]; d[3] = s[3];
        }
        __syncthreads();
        const _Float16* ktb = kt2[0];
        const _Float16* vtb = vt2[0];
#endif

        // ---- S = q^T k for this wave's 16 rows x 32 key-cols ----
        v8f st[2];
#pragma unroll
        for (int t = 0; t < 2; ++t) {
            const _Float16* kr = &ktb[(t * 16 + ln) * KTP];
            H16 kb0, kb1;
            kb0.u4[0] = *(const uint4*)(kr + hh * 8);
            kb0.u4[1] = *(const uint4*)(kr + 16 + hh * 8);
            kb1.u4[0] = *(const uint4*)(kr + 32 + hh * 8);
            kb1.u4[1] = *(const uint4*)(kr + 48 + hh * 8);
            v8f s = {0.f,0.f,0.f,0.f,0.f,0.f,0.f,0.f};
            s = wmma_f16(qa[0], kb0, s);
            s = wmma_f16(qa[1], kb1, s);
            st[t] = s;
        }

        // ---- online softmax (row = i + 8*hh lives in one 16-lane half) ----
        float tm[8];
#pragma unroll
        for (int i = 0; i < 8; ++i) tm[i] = fmaxf(st[0][i], st[1][i]);
#pragma unroll
        for (int m = 1; m <= 8; m <<= 1)
#pragma unroll
            for (int i = 0; i < 8; ++i)
                tm[i] = fmaxf(tm[i], __shfl_xor(tm[i], m, 32));
        float p0[8], p1[8], ps[8], fs[8];
#pragma unroll
        for (int i = 0; i < 8; ++i) {
            const float mn = fmaxf(rm[i], tm[i]);
            fs[i] = __expf(rm[i] - mn);
            p0[i] = __expf(st[0][i] - mn);
            p1[i] = __expf(st[1][i] - mn);
            ps[i] = p0[i] + p1[i];
            rm[i] = mn;
        }
#pragma unroll
        for (int m = 1; m <= 8; m <<= 1)
#pragma unroll
            for (int i = 0; i < 8; ++i) ps[i] += __shfl_xor(ps[i], m, 32);
#pragma unroll
        for (int i = 0; i < 8; ++i) rs[i] = rs[i] * fs[i] + ps[i];
#pragma unroll
        for (int ct = 0; ct < 8; ++ct)
#pragma unroll
            for (int i = 0; i < 8; ++i) acc[ct][i] *= fs[i];

        // ---- P: D-layout -> LDS -> A-layout (same wave; DS is in-order) ----
#pragma unroll
        for (int i = 0; i < 8; ++i) {
            pw[(i + 8 * hh) * PTP + ln]      = (_Float16)p0[i];
            pw[(i + 8 * hh) * PTP + 16 + ln] = (_Float16)p1[i];
        }
        asm volatile("" ::: "memory");   // keep compile-time order of DS ops
        H16 pa;
        pa.u4[0] = *(const uint4*)&pw[ln * PTP + hh * 8];
        pa.u4[1] = *(const uint4*)&pw[ln * PTP + 16 + hh * 8];

        // ---- acc += P * V for this wave's 128 channels ----
#pragma unroll
        for (int ct = 0; ct < 8; ++ct) {
            const int c = hch * 128 + ct * 16 + ln;
            H16 vb;
            vb.u4[0] = *(const uint4*)&vtb[c * VTP + hh * 8];
            vb.u4[1] = *(const uint4*)&vtb[c * VTP + 16 + hh * 8];
            acc[ct] = wmma_f16(pa, vb, acc[ct]);
        }
    }

    // ---- epilogue: out = gamma * acc/rowsum + x ----
    const float g = gamma[0];
    float inv[8];
#pragma unroll
    for (int i = 0; i < 8; ++i) inv[i] = 1.0f / rs[i];
#pragma unroll
    for (int ct = 0; ct < 8; ++ct) {
        const int c = hch * 128 + ct * 16 + ln;
        const size_t base =
            ((size_t)b * CDIM + c) * NPIX + n0 + r * 16 + 8 * hh;
        const float4* xp = (const float4*)(x + base);
        float4* op = (float4*)(out + base);
        float4 x0 = xp[0], x1 = xp[1];
        float4 o0, o1;
        o0.x = g * acc[ct][0] * inv[0] + x0.x;
        o0.y = g * acc[ct][1] * inv[1] + x0.y;
        o0.z = g * acc[ct][2] * inv[2] + x0.z;
        o0.w = g * acc[ct][3] * inv[3] + x0.w;
        o1.x = g * acc[ct][4] * inv[4] + x1.x;
        o1.y = g * acc[ct][5] * inv[5] + x1.y;
        o1.z = g * acc[ct][6] * inv[6] + x1.z;
        o1.w = g * acc[ct][7] * inv[7] + x1.w;
        op[0] = o0; op[1] = o1;
    }
}

// ---------------------------------------------------------------------------
extern "C" void kernel_launch(void* const* d_in, const int* in_sizes, int n_in,
                              void* d_out, int out_size, void* d_ws, size_t ws_size,
                              hipStream_t stream) {
    (void)in_sizes; (void)n_in; (void)out_size; (void)ws_size;
    const float* x     = (const float*)d_in[0];
    const float* Wq    = (const float*)d_in[1];
    const float* bq    = (const float*)d_in[2];
    const float* Wk    = (const float*)d_in[3];
    const float* bk    = (const float*)d_in[4];
    const float* Wv    = (const float*)d_in[5];
    const float* bv    = (const float*)d_in[6];
    const float* gamma = (const float*)d_in[7];
    float* out = (float*)d_out;

    _Float16* qT = (_Float16*)d_ws;                       // 4*4096*64
    _Float16* kT = qT + (size_t)4 * NPIX * CQ;            // 4*4096*64
    _Float16* vF = kT + (size_t)4 * NPIX * CQ;            // 4*256*4096

    dim3 grid(NPIX / 64, 4), block(256);
    qkv_proj_kernel<<<grid, block, 0, stream>>>(x, Wq, bq, Wk, bk, Wv, bv,
                                                qT, kT, vF);
    flash_attn_kernel<<<grid, block, 0, stream>>>(qT, kT, vF, x, gamma, out);
}
